// CrossAttention_11501922419335
// MI455X (gfx1250) — compile-verified, matrix operand order checked by reading, not run on hardware
//
#include <hip/hip_runtime.h>
#include <hip/hip_bf16.h>
#include <stdint.h>

#define DEV __device__ __forceinline__

typedef __attribute__((ext_vector_type(16))) __bf16 bf16x16;
typedef __attribute__((ext_vector_type(8)))  __bf16 bf16x8;
typedef __attribute__((ext_vector_type(4)))  __bf16 bf16x4;
typedef __attribute__((ext_vector_type(8)))  float  f32x8;

static constexpr int BATCH  = 2;
static constexpr int SEQ    = 4096;
static constexpr int HEADS  = 8;
static constexpr int DH     = 40;
static constexpr int DP     = 64;    // head dim padded for WMMA K/N tiling
static constexpr int DMODEL = 320;   // == D_ATTN == N of all projections
static constexpr int DCOND  = 768;
static constexpr float LOG2E = 1.44269504088896340736f;

DEV f32x8 zero8() { f32x8 v; for (int i = 0; i < 8; ++i) v[i] = 0.0f; return v; }

// Wave-relative LDS byte address of a __shared__ object (generic -> AS(3)).
DEV uint32_t lds_off(const void* p) {
  return (uint32_t)(unsigned long long)(__attribute__((address_space(3))) const void*)p;
}

// CDNA5 async memory->LDS copy, 16 bytes per lane, GV addressing mode.
// ISA §15.18: VDST = LDS address VGPR, VADDR = 64-bit global address, SADDR=off.
// Tracked by ASYNCcnt; no VGPR data movement.
DEV void async_b128(uint32_t lds_addr, uint64_t gaddr) {
  asm volatile("global_load_async_to_lds_b128 %0, %1, off"
               :: "v"(lds_addr), "v"(gaddr) : "memory");
}
DEV void wait_async0() {
#if __has_builtin(__builtin_amdgcn_s_wait_asynccnt)
  __builtin_amdgcn_s_wait_asynccnt(0);
#else
  asm volatile("s_wait_asynccnt 0x0" ::: "memory");
#endif
}
// async loads retire in order: waiting <= N with one extra tile (N ops/thread)
// in flight exactly drains the previous tile.
DEV void wait_async4() {   // GEMM tiles: 2 A + 2 B chunks per thread
#if __has_builtin(__builtin_amdgcn_s_wait_asynccnt)
  __builtin_amdgcn_s_wait_asynccnt(4);
#else
  asm volatile("s_wait_asynccnt 0x4" ::: "memory");
#endif
}
DEV void wait_async7() {   // attention KV tiles: K:4 + V:3 chunks per thread
#if __has_builtin(__builtin_amdgcn_s_wait_asynccnt)
  __builtin_amdgcn_s_wait_asynccnt(7);
#else
  asm volatile("s_wait_asynccnt 0x7" ::: "memory");
#endif
}

// Load a 16x32 bf16 A/B WMMA fragment row for this lane from LDS.
// Per ISA layout: elements 0..7 <- k = half*8 .. half*8+7,
//                 elements 8..15 <- k = 16+half*8 .. 16+half*8+7.
DEV bf16x16 ld_frag(const __bf16* rowbase, int half) {
  bf16x8 lo = *(const bf16x8*)(rowbase + half * 8);
  bf16x8 hi = *(const bf16x8*)(rowbase + 16 + half * 8);
  return __builtin_shufflevector(lo, hi, 0,1,2,3,4,5,6,7,8,9,10,11,12,13,14,15);
}

DEV f32x8 wmma_bf16(bf16x16 a, bf16x16 b, f32x8 c) {
  return __builtin_amdgcn_wmma_f32_16x16x32_bf16(false, a, false, b, (short)0, c,
                                                 false, false);
}

// ---------------------------------------------------------------------------
// Prep kernels: one-time f32 -> bf16 conversion of activations, and
// convert+transpose of weights to [N][K] bf16 (so GEMM B tiles are contiguous
// async copies and no per-tile conversion/transposition remains in the loops).
// ---------------------------------------------------------------------------
__global__ __launch_bounds__(256)
void cvt_kernel(const float* __restrict__ src, __bf16* __restrict__ dst, int n4) {
  int i = blockIdx.x * 256 + threadIdx.x;
  if (i < n4) {
    float4 v = ((const float4*)src)[i];
    bf16x4 o;
    o[0] = (__bf16)v.x; o[1] = (__bf16)v.y; o[2] = (__bf16)v.z; o[3] = (__bf16)v.w;
    ((bf16x4*)dst)[i] = o;
  }
}

__global__ __launch_bounds__(256)
void wtrans_kernel(const float* __restrict__ w,   // [Kd][320] row-major
                   __bf16* __restrict__ wt,       // [320][Kd] bf16
                   int Kd) {
  __shared__ float tile[16][17];
  int tx = threadIdx.x & 15, ty = threadIdx.x >> 4;
  int n0 = blockIdx.x * 16, k0 = blockIdx.y * 16;
  tile[ty][tx] = w[(size_t)(k0 + ty) * DMODEL + n0 + tx];
  __syncthreads();
  wt[(size_t)(n0 + ty) * Kd + k0 + tx] = (__bf16)tile[tx][ty];
}

// ---------------------------------------------------------------------------
// Shared GEMM core: 64x64 output tile, all-bf16, async double-buffered tiles.
// A [M][Kd] row-major, Wt [N][Kd] row-major (pre-transposed weights).
// ---------------------------------------------------------------------------
DEV void gemm_tiles(const __bf16* __restrict__ A, const __bf16* __restrict__ Wt,
                    int Kd, int m0, int n0, int tid, int wave, int ln, int half,
                    __bf16 (&As)[2][64 * 40], __bf16 (&Bs)[2][64 * 40],
                    f32x8 (&acc)[4]) {
  const uint32_t as_l[2] = { lds_off(As[0]), lds_off(As[1]) };
  const uint32_t bs_l[2] = { lds_off(Bs[0]), lds_off(Bs[1]) };
  auto issue = [&](int k0, int bufi) {
    int r = tid >> 2, c = tid & 3;           // 64 rows x 4 x 16B chunks each
    for (int i = 0; i < 2; ++i, r += 32) {
      async_b128(as_l[bufi] + r * 80 + c * 16,
                 (uint64_t)(uintptr_t)(A + (size_t)(m0 + r) * Kd + k0 + c * 8));
      async_b128(bs_l[bufi] + r * 80 + c * 16,
                 (uint64_t)(uintptr_t)(Wt + (size_t)(n0 + r) * Kd + k0 + c * 8));
    }
  };
  issue(0, 0);
  int bufi = 0;
  for (int k0 = 0; k0 < Kd; k0 += 32) {
    if (k0 + 32 < Kd) { issue(k0 + 32, bufi ^ 1); wait_async4(); }
    else              { wait_async0(); }
    __syncthreads();
    bf16x16 bfrag = ld_frag(&Bs[bufi][(wave * 16 + ln) * 40], half);
    for (int t = 0; t < 4; ++t) {
      bf16x16 afrag = ld_frag(&As[bufi][(t * 16 + ln) * 40], half);
      acc[t] = wmma_bf16(afrag, bfrag, acc[t]);
    }
    __syncthreads();
    bufi ^= 1;
  }
}

// ---------------------------------------------------------------------------
// Kernel 1: QKV projection (all-bf16 GEMM).
//   Q -> [B,H,S,DP]   (scaled by Dh^-0.5 * log2(e); softmax done in base 2)
//   K -> [B,H,S,DP]
//   V -> [B,H,DP,SEQ] (d-major, so the attention V^T tile is a contiguous
//                      async copy instead of a per-element LDS transpose)
// ---------------------------------------------------------------------------
__global__ __launch_bounds__(128)
void qkv_kernel(const __bf16* __restrict__ Xb, const __bf16* __restrict__ Cb,
                const __bf16* __restrict__ Wqt, const __bf16* __restrict__ Wkt,
                const __bf16* __restrict__ Wvt,
                __bf16* __restrict__ Qp, __bf16* __restrict__ Kp,
                __bf16* __restrict__ Vp) {
  const int which = blockIdx.z;
  const __bf16* A  = (which == 0) ? Xb  : Cb;
  const __bf16* Wt = (which == 0) ? Wqt : (which == 1 ? Wkt : Wvt);
  __bf16* dst      = (which == 0) ? Qp  : (which == 1 ? Kp : Vp);
  const int Kd     = (which == 0) ? DMODEL : DCOND;
  const float fold = (which == 0) ? (0.15811388300841897f * LOG2E) : 1.0f;

  const int n0 = blockIdx.x * 64;
  const int m0 = blockIdx.y * 64;
  const int tid = threadIdx.x;
  const int lane = tid & 31, wave = tid >> 5;
  const int ln = lane & 15, half = lane >> 4;

  __shared__ __align__(16) __bf16 As[2][64 * 40];
  __shared__ __align__(16) __bf16 Bs[2][64 * 40];

  f32x8 acc[4];
  for (int t = 0; t < 4; ++t) acc[t] = zero8();
  gemm_tiles(A, Wt, Kd, m0, n0, tid, wave, ln, half, As, Bs, acc);

  const int ncol = n0 + wave * 16 + ln;
  const int h = ncol / DH, d = ncol % DH;
  for (int t = 0; t < 4; ++t)
    for (int r = 0; r < 8; ++r) {
      int m = m0 + t * 16 + r + 8 * half;
      int b = m >> 12, s = m & (SEQ - 1);
      size_t off = (which == 2)
          ? (((size_t)b * HEADS + h) * DP + d) * SEQ + s      // V: d-major
          : (((size_t)b * HEADS + h) * SEQ + s) * DP + d;     // Q/K: s-major
      dst[off] = (__bf16)(acc[t][r] * fold);
    }
}

// ---------------------------------------------------------------------------
// Kernel 2: flash attention with async-to-LDS double buffering.
// Each block: 64 Q rows of one (b,h); each wave owns 16 rows; KV streamed in
// 64-row tiles fetched with global_load_async_to_lds_b128, one tile ahead.
// ---------------------------------------------------------------------------
__global__ __launch_bounds__(128)
void attn_kernel(const __bf16* __restrict__ Qp, const __bf16* __restrict__ Kp,
                 const __bf16* __restrict__ Vp, __bf16* __restrict__ Ob) {
  const int q0 = blockIdx.x * 64;
  const int h = blockIdx.y, b = blockIdx.z;
  const int tid = threadIdx.x;
  const int lane = tid & 31, wave = tid >> 5;
  const int ln = lane & 15, half = lane >> 4;
  const size_t bh  = ((size_t)b * HEADS + h) * SEQ;   // Q/K row base
  const size_t bhd = ((size_t)b * HEADS + h) * DP;    // V (d-major) row base

  __shared__ __align__(16) __bf16 Qs[64 * 72];         // [qrow][d]
  __shared__ __align__(16) __bf16 Ks[2][64 * 72];      // [kvrow][d] (B^T layout)
  __shared__ __align__(16) __bf16 Vt[2][48 * 72];      // [d][kvrow] (already ^T)
  __shared__ __align__(16) __bf16 Pb[4][16 * 72];      // per-wave P staging

  const uint32_t qs_l  = lds_off(Qs);
  const uint32_t ks_l0 = lds_off(Ks[0]), ks_l1 = lds_off(Ks[1]);
  const uint32_t vt_l0 = lds_off(Vt[0]), vt_l1 = lds_off(Vt[1]);

  auto issue_tile = [&](int kv0, int bufi) {
    const uint32_t kl = bufi ? ks_l1 : ks_l0;
    const uint32_t vl = bufi ? vt_l1 : vt_l0;
    int row = tid >> 3, g = tid & 7;
    for (int i = 0; i < 4; ++i, row += 16)
      async_b128(kl + row * 144 + g * 16,
                 (uint64_t)(uintptr_t)(Kp + (bh + kv0 + row) * DP + g * 8));
    for (int task = tid; task < 48 * 8; task += 128) {
      int d = task >> 3, gg = task & 7;
      async_b128(vl + d * 144 + gg * 16,
                 (uint64_t)(uintptr_t)(Vp + (bhd + d) * SEQ + kv0 + gg * 8));
    }
  };

  {  // async-load Q tile (4 chunks/thread) and first KV tile
    int r = tid >> 3, g = tid & 7;
    for (int i = 0; i < 4; ++i, r += 16)
      async_b128(qs_l + r * 144 + g * 16,
                 (uint64_t)(uintptr_t)(Qp + (bh + q0 + r) * DP + g * 8));
  }
  issue_tile(0, 0);
  wait_async7();     // Q (oldest 4 ops) complete; tile0 may still be in flight
  __syncthreads();

  bf16x16 qfrag[2];
  for (int ks = 0; ks < 2; ++ks)
    qfrag[ks] = ld_frag(&Qs[(wave * 16 + ln) * 72 + ks * 32], half);

  f32x8 oacc[3];
  for (int u = 0; u < 3; ++u) oacc[u] = zero8();
  float mi[8], li[8];
  for (int r = 0; r < 8; ++r) { mi[r] = -1e30f; li[r] = 0.0f; }

  int bufi = 0;
  for (int kv0 = 0; kv0 < SEQ; kv0 += 64) {
    if (kv0 + 64 < SEQ) { issue_tile(kv0 + 64, bufi ^ 1); wait_async7(); }
    else                { wait_async0(); }
    __syncthreads();   // publish this tile's LDS to all waves

    const __bf16* Kb = Ks[bufi];
    const __bf16* Vb = Vt[bufi];

    // S = Q K^T over this 64-KV tile (pre-scaled, log2-domain logits)
    f32x8 sacc[4];
    for (int t = 0; t < 4; ++t) sacc[t] = zero8();
    for (int ks = 0; ks < 2; ++ks)
      for (int t = 0; t < 4; ++t) {
        bf16x16 kf = ld_frag(&Kb[(t * 16 + ln) * 72 + ks * 32], half);
        sacc[t] = wmma_bf16(qfrag[ks], kf, sacc[t]);
      }

    // online softmax (exp2): row max across 4 N-tiles + 16 lanes
    float alpha[8];
    for (int r = 0; r < 8; ++r) {
      float v = fmaxf(fmaxf(sacc[0][r], sacc[1][r]), fmaxf(sacc[2][r], sacc[3][r]));
      for (int off = 1; off < 16; off <<= 1) v = fmaxf(v, __shfl_xor(v, off, 32));
      float mnew = fmaxf(mi[r], v);
      alpha[r] = exp2f(mi[r] - mnew);
      mi[r] = mnew;
    }
    float rs[8];
    for (int r = 0; r < 8; ++r) rs[r] = 0.0f;
    for (int t = 0; t < 4; ++t)
      for (int r = 0; r < 8; ++r) {
        float p = exp2f(sacc[t][r] - mi[r]);
        sacc[t][r] = p;
        rs[r] += p;
      }
    for (int r = 0; r < 8; ++r) {
      for (int off = 1; off < 16; off <<= 1) rs[r] += __shfl_xor(rs[r], off, 32);
      li[r] = li[r] * alpha[r] + rs[r];
    }
    for (int u = 0; u < 3; ++u)
      for (int r = 0; r < 8; ++r) oacc[u][r] *= alpha[r];

    // stage P (C layout -> A layout) via per-wave LDS region (DScnt-ordered)
    for (int t = 0; t < 4; ++t)
      for (int r = 0; r < 8; ++r)
        Pb[wave][(r + 8 * half) * 72 + t * 16 + ln] = (__bf16)sacc[t][r];

    // O += P @ V
    for (int ks = 0; ks < 2; ++ks) {
      bf16x16 pf = ld_frag(&Pb[wave][ln * 72 + ks * 32], half);
      for (int u = 0; u < 3; ++u) {
        bf16x16 vf = ld_frag(&Vb[(u * 16 + ln) * 72 + ks * 32], half);
        oacc[u] = wmma_bf16(pf, vf, oacc[u]);
      }
    }
    __syncthreads();   // all waves done with buf before it is refilled
    bufi ^= 1;
  }

  for (int r = 0; r < 8; ++r) li[r] = 1.0f / li[r];
  for (int u = 0; u < 3; ++u) {
    int dd = u * 16 + ln;
    if (dd < DH)
      for (int r = 0; r < 8; ++r) {
        int m = q0 + wave * 16 + r + 8 * half;
        Ob[((size_t)b * SEQ + m) * DMODEL + h * DH + dd] =
            (__bf16)(oacc[u][r] * li[r]);
      }
  }
}

// ---------------------------------------------------------------------------
// Kernel 3: output projection  out[8192,320] = Ob(bf16) @ Wo^T(bf16) + bo
// ---------------------------------------------------------------------------
__global__ __launch_bounds__(128)
void oproj_kernel(const __bf16* __restrict__ Ob, const __bf16* __restrict__ Wot,
                  const float* __restrict__ bo, float* __restrict__ out) {
  const int n0 = blockIdx.x * 64;
  const int m0 = blockIdx.y * 64;
  const int tid = threadIdx.x;
  const int lane = tid & 31, wave = tid >> 5;
  const int ln = lane & 15, half = lane >> 4;

  __shared__ __align__(16) __bf16 As[2][64 * 40];
  __shared__ __align__(16) __bf16 Bs[2][64 * 40];

  f32x8 acc[4];
  for (int t = 0; t < 4; ++t) acc[t] = zero8();
  gemm_tiles(Ob, Wot, DMODEL, m0, n0, tid, wave, ln, half, As, Bs, acc);

  const int n = n0 + wave * 16 + ln;
  const float bias = bo[n];
  for (int t = 0; t < 4; ++t)
    for (int r = 0; r < 8; ++r) {
      int m = m0 + t * 16 + r + 8 * half;
      out[(size_t)m * DMODEL + n] = acc[t][r] + bias;
    }
}

// ---------------------------------------------------------------------------
extern "C" void kernel_launch(void* const* d_in, const int* in_sizes, int n_in,
                              void* d_out, int out_size, void* d_ws, size_t ws_size,
                              hipStream_t stream) {
  const float* x    = (const float*)d_in[0];
  const float* cond = (const float*)d_in[1];
  const float* Wq   = (const float*)d_in[2];
  const float* Wk   = (const float*)d_in[3];
  const float* Wv   = (const float*)d_in[4];
  const float* Wo   = (const float*)d_in[5];
  const float* bo   = (const float*)d_in[6];
  float* out        = (float*)d_out;

  const int MROWS = BATCH * SEQ;                     // 8192
  char* ws = (char*)d_ws;
  const size_t qkv_bytes = (size_t)BATCH * HEADS * SEQ * DP * sizeof(__bf16);
  __bf16* Qp  = (__bf16*)(ws);                        ws += qkv_bytes;
  __bf16* Kp  = (__bf16*)(ws);                        ws += qkv_bytes;
  __bf16* Vp  = (__bf16*)(ws);                        ws += qkv_bytes;
  __bf16* Ob  = (__bf16*)(ws);  ws += (size_t)MROWS * DMODEL * sizeof(__bf16);
  __bf16* Xb  = (__bf16*)(ws);  ws += (size_t)MROWS * DMODEL * sizeof(__bf16);
  __bf16* Cb  = (__bf16*)(ws);  ws += (size_t)MROWS * DCOND  * sizeof(__bf16);
  __bf16* Wqt = (__bf16*)(ws);  ws += (size_t)DMODEL * DMODEL * sizeof(__bf16);
  __bf16* Wkt = (__bf16*)(ws);  ws += (size_t)DMODEL * DCOND  * sizeof(__bf16);
  __bf16* Wvt = (__bf16*)(ws);  ws += (size_t)DMODEL * DCOND  * sizeof(__bf16);
  __bf16* Wot = (__bf16*)(ws);  ws += (size_t)DMODEL * DMODEL * sizeof(__bf16);

  // zero the Dh->DP pad lanes of Q/K/V (graph-capturable async memset)
  hipMemsetAsync(Qp, 0, 3 * qkv_bytes, stream);

  // one-time bf16 conversion / weight transposition
  cvt_kernel<<<(MROWS * DMODEL / 4 + 255) / 256, 256, 0, stream>>>(
      x, Xb, MROWS * DMODEL / 4);
  cvt_kernel<<<(MROWS * DCOND / 4 + 255) / 256, 256, 0, stream>>>(
      cond, Cb, MROWS * DCOND / 4);
  wtrans_kernel<<<dim3(DMODEL / 16, DMODEL / 16), 256, 0, stream>>>(Wq, Wqt, DMODEL);
  wtrans_kernel<<<dim3(DMODEL / 16, DCOND / 16), 256, 0, stream>>>(Wk, Wkt, DCOND);
  wtrans_kernel<<<dim3(DMODEL / 16, DCOND / 16), 256, 0, stream>>>(Wv, Wvt, DCOND);
  wtrans_kernel<<<dim3(DMODEL / 16, DMODEL / 16), 256, 0, stream>>>(Wo, Wot, DMODEL);

  qkv_kernel<<<dim3(DMODEL / 64, MROWS / 64, 3), 128, 0, stream>>>(
      Xb, Cb, Wqt, Wkt, Wvt, Qp, Kp, Vp);
  attn_kernel<<<dim3(SEQ / 64, HEADS, BATCH), 128, 0, stream>>>(Qp, Kp, Vp, Ob);
  oproj_kernel<<<dim3(DMODEL / 64, MROWS / 64), 128, 0, stream>>>(Ob, Wot, bo, out);
}
